// Upsampling_attr_70643622085270
// MI455X (gfx1250) — matrix operand order repaired and checked by reading, not compiled, not run on hardware
//
#include <hip/hip_runtime.h>

// ---------------------------------------------------------------------------
// MI455X (gfx1250, wave32) implementation of the sparse-conv upsampling net.
// Every _sconv/_lin is a gathered GEMM on v_wmma_f32_16x16x32_bf16.
//  - activations: padded f32 + bf16 mirrors (pad channels always zero)
//  - weights packed per-call into native B-fragment layout; per-tap slabs
//    streamed into double-buffered LDS with GLOBAL_LOAD_ASYNC_TO_LDS_B128
//    (ASYNCcnt) and shared by the 8 waves of a block
//  - each wave computes MT*16 rows x COP cols, sharing B fragments across
//    the MT row-tiles (MT=2 for COP<=64, MT=1 for COP=128)
//  - A fragments gathered per-lane with sentinel -> zero rows
// ---------------------------------------------------------------------------

typedef __bf16 bf16_t;
typedef __attribute__((ext_vector_type(16))) __bf16 v16bf;
typedef __attribute__((ext_vector_type(8)))  __bf16 v8bf;
typedef __attribute__((ext_vector_type(8)))  float  v8f;

union BF16x16 { v16bf v; v8bf h[2]; };

__device__ inline bf16_t f2bf(float f) {
  unsigned u = __builtin_bit_cast(unsigned, f);
  unsigned r = u + 0x7FFFu + ((u >> 16) & 1u);   // round-to-nearest-even
  unsigned short hh = (unsigned short)(r >> 16);
  return __builtin_bit_cast(bf16_t, hh);
}

__device__ inline v8bf v8bf_zero() {
  unsigned short z = 0;
  bf16_t zb = __builtin_bit_cast(bf16_t, z);
  v8bf r;
#pragma unroll
  for (int i = 0; i < 8; ++i) r[i] = zb;
  return r;
}

// ---------------------------------------------------------------------------
// Gathered-GEMM:  out[n, COP] = sum_k gather_k(xb)[n, CIP] @ Wk[CIP, COP] + b
//   NN = COP/16 column chunks; MT = row tiles of 16 per wave.
//   A fragment (16-bit, 16x32): lane half hl: j<8 -> K=cc*32+hl*16+j,
//                               j>=8 -> K=cc*32+hl*16+j
//   B fragment (32x16): lane L -> col nn*16+(L&15), K = (L>>4)*16 + j
// ---------------------------------------------------------------------------
template<int NN, int MT>
__global__ void __launch_bounds__(256)
sconv_gemm(const bf16_t* __restrict__ xb, int cip, int nSrc,
           const int* __restrict__ nbr, int K, int nCC,
           const bf16_t* __restrict__ wpk, const float* __restrict__ bp,
           float* __restrict__ outf, bf16_t* __restrict__ outb,
           int n, int doRelu)
{
  extern __shared__ char smem[];
  // LDS aperture keeps the wave-relative byte offset in addr[31:0] (ISA 10.2)
  const unsigned ldsBase = (unsigned)(size_t)smem;

  const int tid  = threadIdx.x;
  const int lane = tid & 31;
  const int wave = tid >> 5;
  const int hl   = lane >> 4;        // half-wave select
  const int l15  = lane & 15;
  const int tile = blockIdx.x * 8 + wave;
  const int row0 = tile * (MT * 16);

  int myRow[MT], rcl[MT];
#pragma unroll
  for (int m = 0; m < MT; ++m) {
    myRow[m] = row0 + m * 16 + l15;
    rcl[m] = (myRow[m] < n) ? myRow[m] : (n - 1);
  }

  v8f acc[MT][NN];
  for (int m = 0; m < MT; ++m)
    for (int i = 0; i < NN; ++i)
      for (int e = 0; e < 8; ++e) acc[m][i][e] = 0.f;

  const int slabInt4 = nCC * NN * 64;            // 16B chunks per tap
  const unsigned slabBytes = (unsigned)slabInt4 * 16u;

  // --- async stage of one tap's weight slab into LDS buffer b ------------
  auto stage = [&](int k, int b) {
    const char* src = (const char*)wpk + (size_t)k * slabBytes;
    unsigned dstBase = ldsBase + (unsigned)b * slabBytes;
    for (int i = tid; i < slabInt4; i += 256) {
      unsigned lds = dstBase + (unsigned)i * 16u;
      const char* g = src + (size_t)i * 16;
      asm volatile("global_load_async_to_lds_b128 %0, %1, off"
                   :: "v"(lds), "v"(g) : "memory");
    }
  };

  // prologue: stage tap 0
  stage(0, 0);
  asm volatile("s_wait_asynccnt 0x0" ::: "memory");
  __syncthreads();

  for (int k = 0; k < K; ++k) {
    const int cur = k & 1;
    if (k + 1 < K) stage(k + 1, cur ^ 1);      // overlap with compute

    int idx[MT];
    bool valid[MT];
#pragma unroll
    for (int m = 0; m < MT; ++m) {
      idx[m] = nbr ? nbr[(size_t)k * n + rcl[m]] : myRow[m];
      valid[m] = (idx[m] < nSrc) && (myRow[m] < n);
    }
    if (nbr && (k + 1 < K))
      __builtin_prefetch(&nbr[(size_t)(k + 1) * n + rcl[0]], 0, 0);

    const char* bufc = smem + (size_t)cur * slabBytes;

    for (int cc = 0; cc < nCC; ++cc) {
      BF16x16 a[MT];
#pragma unroll
      for (int m = 0; m < MT; ++m) {
        if (valid[m]) {
          const bf16_t* arow = xb + (size_t)idx[m] * cip + cc * 32;
          a[m].h[0] = *(const v8bf*)(arow + hl * 16);
          a[m].h[1] = *(const v8bf*)(arow + hl * 16 + 8);
        } else {
          a[m].h[0] = v8bf_zero();
          a[m].h[1] = v8bf_zero();
        }
      }
      const char* bbase = bufc + (size_t)(cc * NN) * 1024 + (size_t)lane * 32;
#pragma unroll
      for (int nn = 0; nn < NN; ++nn) {
        BF16x16 b;
        b.h[0] = *(const v8bf*)(bbase + (size_t)nn * 1024);
        b.h[1] = *(const v8bf*)(bbase + (size_t)nn * 1024 + 16);
#pragma unroll
        for (int m = 0; m < MT; ++m)
          acc[m][nn] = __builtin_amdgcn_wmma_f32_16x16x32_bf16(
              false, a[m].v, false, b.v, (short)0, acc[m][nn], false, false);
      }
    }

    // next tap's slab must be fully in LDS before anyone reads it
    asm volatile("s_wait_asynccnt 0x0" ::: "memory");
    __syncthreads();
  }

  // epilogue: bias (+relu); D layout: VGPR e -> row row0+m*16+hl*8+e,
  // col nn*16+l15.
  if (row0 < n) {
#pragma unroll
    for (int nn = 0; nn < NN; ++nn) {
      const int col = nn * 16 + l15;
      const float bias = bp[col];
#pragma unroll
      for (int m = 0; m < MT; ++m) {
#pragma unroll
        for (int e = 0; e < 8; ++e) {
          int r = row0 + m * 16 + hl * 8 + e;
          if (r < n) {
            float v = acc[m][nn][e] + bias;
            if (doRelu) v = v > 0.f ? v : 0.f;
            size_t o = (size_t)r * (NN * 16) + col;
            outf[o] = v;
            if (outb) outb[o] = f2bf(v);
          }
        }
      }
    }
  }
}

// Pack W[k][ci][co] (f32) into per-tap B-fragment slabs (bf16), zero-padded.
__global__ void pack_w(const float* __restrict__ W, bf16_t* __restrict__ wpk,
                       int K, int CI, int CO, int nCC, int NN)
{
  int t = blockIdx.x * blockDim.x + threadIdx.x;
  int total = K * nCC * NN * 32;
  if (t >= total) return;
  int lane = t & 31, blk = t >> 5;
  int nn = blk % NN;
  int tmp = blk / NN;
  int cc = tmp % nCC;
  int k  = tmp / nCC;
  int hl = lane >> 4;
  int col = nn * 16 + (lane & 15);
  bf16_t* dst = wpk + (size_t)t * 16;
  for (int j = 0; j < 16; ++j) {
    int ci = cc * 32 + hl * 16 + j;
    float v = 0.f;
    if (ci < CI && col < CO) v = W[((size_t)k * CI + ci) * CO + col];
    dst[j] = f2bf(v);
  }
}

__global__ void pack_bias(const float* __restrict__ b, float* __restrict__ bp,
                          int CO, int COP)
{
  int c = threadIdx.x;
  if (c < COP) bp[c] = (c < CO) ? b[c] : 0.f;
}

__global__ void pad_convert(const float* __restrict__ x, bf16_t* __restrict__ xb,
                            int n, int C, int CP)
{
  size_t t = (size_t)blockIdx.x * blockDim.x + threadIdx.x;
  if (t >= (size_t)n * CP) return;
  int ch = (int)(t % CP);
  size_t row = t / CP;
  xb[t] = f2bf(ch < C ? x[row * C + ch] : 0.f);
}

// y = relu(concat(out0, out1) + x), writes padded f32 + bf16 (pad ch -> 0)
__global__ void res_concat_relu(const float* __restrict__ o0, int s0,
                                const float* __restrict__ o1, int s1,
                                const float* __restrict__ x, int sx,
                                float* __restrict__ yf, bf16_t* __restrict__ yb,
                                int n, int C, int CP)
{
  size_t t = (size_t)blockIdx.x * blockDim.x + threadIdx.x;
  if (t >= (size_t)n * CP) return;
  int ch = (int)(t % CP);
  size_t row = t / CP;
  float v = 0.f;
  if (ch < C) {
    int h = C >> 1;
    float o = (ch < h) ? o0[row * (size_t)s0 + ch]
                       : o1[row * (size_t)s1 + (ch - h)];
    v = o + x[row * (size_t)sx + ch];
    v = v > 0.f ? v : 0.f;
  }
  yf[t] = v;
  yb[t] = f2bf(v);
}

__global__ void strip_out(const float* __restrict__ src, float* __restrict__ o,
                          int n, int CP, int C)
{
  size_t t = (size_t)blockIdx.x * blockDim.x + threadIdx.x;
  if (t >= (size_t)n * C) return;
  int ch = (int)(t % C);
  size_t row = t / C;
  o[t] = src[row * (size_t)CP + ch];
}

// ---------------------------------------------------------------------------

extern "C" void kernel_launch(void* const* d_in, const int* in_sizes, int n_in,
                              void* d_out, int out_size, void* d_ws, size_t ws_size,
                              hipStream_t stream)
{
  (void)n_in; (void)ws_size;
  const int NI = in_sizes[0] / 3;   // 50000
  const int NO = out_size / 3;      // 250000

  const float* X    = (const float*)d_in[0];
  const int* nbr5   = (const int*)d_in[1];
  const int* nbr3t  = (const int*)d_in[2];
  const int* nbr3o  = (const int*)d_in[3];
  auto Wp = [&](int i) { return (const float*)d_in[i]; };

  char* ws = (char*)d_ws;
  size_t off = 0;
  auto alloc = [&](size_t bytes) -> char* {
    char* p = ws + off;
    off += (bytes + 255) & ~(size_t)255;
    return p;
  };

  bf16_t* xb = (bf16_t*)alloc((size_t)NI * 32 * 2);
  float*  Af = (float*) alloc((size_t)NI * 128 * 4);
  bf16_t* Ab = (bf16_t*)alloc((size_t)NI * 128 * 2);
  float*  Bf = (float*) alloc((size_t)NI * 128 * 4);
  bf16_t* Bb = (bf16_t*)alloc((size_t)NI * 128 * 2);
  float*  Pf = (float*) alloc((size_t)NI * 32 * 4);
  bf16_t* Pb = (bf16_t*)alloc((size_t)NI * 32 * 2);
  float*  Qf = (float*) alloc((size_t)NI * 32 * 4);
  bf16_t* Qb = (bf16_t*)alloc((size_t)NI * 32 * 2);
  float*  Rf = (float*) alloc((size_t)NI * 32 * 4);
  bf16_t* Rb = (bf16_t*)alloc((size_t)NI * 32 * 2);
  float*  T0f = (float*) alloc((size_t)NO * 64 * 4);
  bf16_t* T0b = (bf16_t*)alloc((size_t)NO * 64 * 2);
  float*  T1f = (float*) alloc((size_t)NO * 32 * 4);
  bf16_t* T1b = (bf16_t*)alloc((size_t)NO * 32 * 2);
  float*  T2f = (float*) alloc((size_t)NO * 32 * 4);
  bf16_t* wpk = (bf16_t*)alloc((size_t)125 * 64 * 128 * 2);  // max packed layer
  float*  bp  = (float*) alloc(128 * 4);

  auto gemm = [&](const bf16_t* inb, int cip, int nSrc, const int* nbr, int K,
                  int wi, int CI, int CO, float* of, bf16_t* ob, int n, bool relu)
  {
    const int COP = ((CO + 31) / 32) * 32;
    const int NN  = COP / 16;
    const int nCC = cip / 32;
    int total = K * nCC * NN * 32;
    pack_w<<<(total + 255) / 256, 256, 0, stream>>>(Wp(wi), wpk, K, CI, CO, nCC, NN);
    pack_bias<<<1, 128, 0, stream>>>(Wp(wi + 1), bp, CO, COP);
    size_t smem = 2 * (size_t)nCC * NN * 1024;       // double-buffered slabs
    const int rlu = relu ? 1 : 0;
    if (NN == 8) {           // MT=1
      int tiles = (n + 15) / 16, blocks = (tiles + 7) / 8;
      sconv_gemm<8, 1><<<blocks, 256, smem, stream>>>(inb, cip, nSrc, nbr, K, nCC, wpk, bp, of, ob, n, rlu);
    } else if (NN == 4) {    // MT=2
      int tiles = (n + 31) / 32, blocks = (tiles + 7) / 8;
      sconv_gemm<4, 2><<<blocks, 256, smem, stream>>>(inb, cip, nSrc, nbr, K, nCC, wpk, bp, of, ob, n, rlu);
    } else {                 // NN==2, MT=2
      int tiles = (n + 31) / 32, blocks = (tiles + 7) / 8;
      sconv_gemm<2, 2><<<blocks, 256, smem, stream>>>(inb, cip, nSrc, nbr, K, nCC, wpk, bp, of, ob, n, rlu);
    }
  };

  auto ew = [&](size_t elems) { return (unsigned)((elems + 255) / 256); };

  // input: pad 3 -> 32 channels, f32 -> bf16
  pad_convert<<<ew((size_t)NI * 32), 256, 0, stream>>>(X, xb, NI, 3, 32);

  // ---- stage 0 (c=16) --------------------------------------------------
  gemm(xb, 32, NI, nbr5, 125,  4,  3, 16, Af, Ab, NI, false);   // conv0
  gemm(Ab, 32, NI, nullptr, 1, 6, 16,  4, Pf, Pb, NI, true);    // c00
  gemm(Pb, 32, NI, nbr5, 125,  8,  4,  4, Qf, Qb, NI, true);    // c01
  gemm(Qb, 32, NI, nullptr, 1,10,  4,  8, Pf, Pb, NI, true);    // c02 -> out0
  gemm(Ab, 32, NI, nbr5, 125, 12, 16,  4, Qf, Qb, NI, true);    // c10
  gemm(Qb, 32, NI, nbr5, 125, 14,  4,  8, Rf, Rb, NI, true);    // c11 -> out1
  res_concat_relu<<<ew((size_t)NI * 32), 256, 0, stream>>>(Pf, 32, Rf, 32, Af, 32, Bf, Bb, NI, 16, 32);

  // ---- stage 1 (c=32) --------------------------------------------------
  gemm(Bb, 32, NI, nbr5, 125, 16, 16, 32, Af, Ab, NI, false);   // conv1
  gemm(Ab, 32, NI, nullptr, 1,18, 32,  8, Pf, Pb, NI, true);
  gemm(Pb, 32, NI, nbr5, 125, 20,  8,  8, Qf, Qb, NI, true);
  gemm(Qb, 32, NI, nullptr, 1,22,  8, 16, Pf, Pb, NI, true);
  gemm(Ab, 32, NI, nbr5, 125, 24, 32,  8, Qf, Qb, NI, true);
  gemm(Qb, 32, NI, nbr5, 125, 26,  8, 16, Rf, Rb, NI, true);
  res_concat_relu<<<ew((size_t)NI * 32), 256, 0, stream>>>(Pf, 32, Rf, 32, Af, 32, Bf, Bb, NI, 32, 32);

  // ---- stage 2 (c=64) --------------------------------------------------
  gemm(Bb, 32, NI, nbr5, 125, 28, 32, 64, Af, Ab, NI, false);   // conv2 (stride 64)
  gemm(Ab, 64, NI, nullptr, 1,30, 64, 16, Pf, Pb, NI, true);
  gemm(Pb, 32, NI, nbr5, 125, 32, 16, 16, Qf, Qb, NI, true);
  gemm(Qb, 32, NI, nullptr, 1,34, 16, 32, Pf, Pb, NI, true);
  gemm(Ab, 64, NI, nbr5, 125, 36, 64, 16, Qf, Qb, NI, true);
  gemm(Qb, 32, NI, nbr5, 125, 38, 16, 32, Rf, Rb, NI, true);
  res_concat_relu<<<ew((size_t)NI * 64), 256, 0, stream>>>(Pf, 32, Rf, 32, Af, 64, Bf, Bb, NI, 64, 64);

  // ---- head ------------------------------------------------------------
  gemm(Bb, 64, NI, nbr5, 125, 40, 64, 128, Af, Ab, NI, false);  // conv3 (stride 128)
  gemm(Ab, 128, NI, nbr3t, 27, 42, 128, 64, T0f, T0b, NO, false); // target
  gemm(T0b, 64, NO, nbr3o, 27, 44, 64, 32, T1f, T1b, NO, true);   // upout
  gemm(T1b, 32, NO, nbr3o, 27, 46, 32, 3, T2f, nullptr, NO, false); // convout

  strip_out<<<ew((size_t)NO * 3), 256, 0, stream>>>(T2f, (float*)d_out, NO, 32, 3);
}
